// TransformerEncoderLayer_54589034332988
// MI455X (gfx1250) — matrix operand order repaired and checked by reading, not compiled, not run on hardware
//
#include <hip/hip_runtime.h>

// ---------------------------------------------------------------------------
// Types & helpers
// ---------------------------------------------------------------------------
typedef __bf16 bf16_t;
typedef __attribute__((ext_vector_type(16))) __bf16 v16bf;
typedef __attribute__((ext_vector_type(8)))  float  v8f;

__device__ __forceinline__ bf16_t f2bf(float f) {
    unsigned u = __builtin_bit_cast(unsigned, f);
    unsigned r = u + 0x7FFFu + ((u >> 16) & 1u);   // round-to-nearest-even
    unsigned short h = (unsigned short)(r >> 16);
    return __builtin_bit_cast(bf16_t, h);
}

// K-index pattern for 16-bit A fragment (16x32), per ISA 7.12.2 table.
__device__ __forceinline__ int akpat(int j, int half) {
    return ((j & 4) << 2) + (half << 3) + ((j & 3) << 1);
}
// K-index pattern for 16-bit B fragment (32x16): same table order as A
// (VGPR j<4: K = 8*half + 2j; j>=4: K = 16 + 8*half + 2(j-4)).
__device__ __forceinline__ int bkpat(int j, int half) {
    return ((j & 4) << 2) + (half << 3) + ((j & 3) << 1);
}

__device__ __forceinline__ v8f wmma_bf16(v16bf a, v16bf b, v8f c) {
    return __builtin_amdgcn_wmma_f32_16x16x32_bf16(
        false, a, false, b, (short)0, c, false, false);
}

// gfx1250 async global->LDS copy (16B per lane), ASYNCcnt tracked.
__device__ __forceinline__ void async_copy_b128(unsigned lds_off, const void* gaddr) {
    asm volatile("global_load_async_to_lds_b128 %0, %1, off"
                 :: "v"(lds_off), "v"(gaddr)
                 : "memory");
}

// Load 16x32 A fragment from row-major bf16 LDS tile (leading dim lda elems).
__device__ __forceinline__ v16bf load_frag_a(const bf16_t* base, int lda,
                                             int row0, int lane) {
    const int m = row0 + (lane & 15);
    const int half = lane >> 4;
    v16bf f;
#pragma unroll
    for (int j = 0; j < 8; ++j) {
        const int k = akpat(j, half);
        f[2 * j]     = base[m * lda + k];
        f[2 * j + 1] = base[m * lda + k + 1];
    }
    return f;
}
// Load 32x16 B fragment from N-major (col-major, [N][K]) bf16 LDS tile.
// Per lane: two runs of 8 contiguous bf16 -> two ds_load_b128.
__device__ __forceinline__ v16bf load_frag_b_nmajor(const bf16_t* base, int ldk,
                                                    int col0, int lane) {
    const int n = col0 + (lane & 15);
    const int half = lane >> 4;
    v16bf f;
#pragma unroll
    for (int j = 0; j < 8; ++j) {
        const int k = bkpat(j, half);
        f[2 * j]     = base[n * ldk + k];
        f[2 * j + 1] = base[n * ldk + k + 1];
    }
    return f;
}

// ---------------------------------------------------------------------------
// Elementwise f32 -> bf16 conversion (4 elems/thread)
// ---------------------------------------------------------------------------
__global__ __launch_bounds__(256) void cvt_kernel(const float* __restrict__ in,
                                                  bf16_t* __restrict__ out,
                                                  int n) {
    const int i = (blockIdx.x * 256 + threadIdx.x) * 4;
    if (i + 3 < n) {
#pragma unroll
        for (int j = 0; j < 4; ++j) out[i + j] = f2bf(in[i + j]);
    }
}

// ---------------------------------------------------------------------------
// Transposing f32 -> bf16 convert: in [K,N] row-major -> out [N,K] row-major.
// 64x64 LDS tile per block, grid (K/64, N/64).
// ---------------------------------------------------------------------------
__global__ __launch_bounds__(256) void cvt_t_kernel(const float* __restrict__ in,
                                                    bf16_t* __restrict__ out,
                                                    int K, int N) {
    __shared__ bf16_t tile[64][65];
    const int k0 = blockIdx.x * 64;
    const int n0 = blockIdx.y * 64;
    const int t = threadIdx.x;
    const int r = t >> 2, c = (t & 3) * 16;
    const float* src = in + (size_t)(k0 + r) * N + n0 + c;
#pragma unroll
    for (int i = 0; i < 16; ++i) tile[r][c + i] = f2bf(src[i]);
    __syncthreads();
    bf16_t* dst = out + (size_t)(n0 + r) * K + k0 + c;
#pragma unroll
    for (int i = 0; i < 16; ++i) dst[i] = tile[c + i][r];
}

// ---------------------------------------------------------------------------
// bf16 transpose of V head slices: v [B*S, 512] -> vt [(b*8+h)*64 + d][S]
// 64x64 tile per block, grid (S/64, 16).
// ---------------------------------------------------------------------------
__global__ __launch_bounds__(256) void vtrans_kernel(const bf16_t* __restrict__ v,
                                                     bf16_t* __restrict__ vt,
                                                     int S) {
    __shared__ bf16_t tile[64][65];
    const int bh = blockIdx.y;            // 0..15
    const int b = bh >> 3, h = bh & 7;
    const int s0 = blockIdx.x * 64;
    const int t = threadIdx.x;
    const int r = t >> 2, c = (t & 3) * 16;
    const bf16_t* src = v + (size_t)(b * S + s0 + r) * 512 + h * 64 + c;
#pragma unroll
    for (int i = 0; i < 16; ++i) tile[r][c + i] = src[i];
    __syncthreads();
    bf16_t* dst = vt + ((size_t)bh * 64 + r) * S + s0 + c;
#pragma unroll
    for (int i = 0; i < 16; ++i) dst[i] = tile[c + i][r];
}

// ---------------------------------------------------------------------------
// GEMM: C[M,N] = act(A[M,K] @ W[K,N] + bias); A bf16 [M,K], Wt bf16 [N,K].
// Block tile 128x64, BK=32, 256 threads = 8 waves (4x2), wave tile 32x32.
// Double-buffered async global->LDS pipeline (3 copies/wave/stage).
// ---------------------------------------------------------------------------
__device__ __forceinline__ void gemm_stage(const bf16_t* __restrict__ A,
                                           const bf16_t* __restrict__ Wt,
                                           unsigned as_b, unsigned bs_b,
                                           int bm, int bn, int K, int k0,
                                           int tid) {
    // A tile: 128 rows x 32 k = 8192 B, 4 x 16B chunks per row
    int c = tid, r = c >> 2, q = c & 3;
    async_copy_b128(as_b + (unsigned)(r * 64 + q * 16),
                    A + (size_t)(bm + r) * K + k0 + q * 8);
    c = tid + 256; r = c >> 2; q = c & 3;
    async_copy_b128(as_b + (unsigned)(r * 64 + q * 16),
                    A + (size_t)(bm + r) * K + k0 + q * 8);
    // B tile (N-major): 64 rows x 32 k = 4096 B
    r = tid >> 2; q = tid & 3;
    async_copy_b128(bs_b + (unsigned)(r * 64 + q * 16),
                    Wt + (size_t)(bn + r) * K + k0 + q * 8);
}

template <bool OUT_BF16, bool RELU>
__global__ __launch_bounds__(256) void gemm_kernel(
    const bf16_t* __restrict__ A, const bf16_t* __restrict__ Wt,
    const float* __restrict__ bias, void* __restrict__ C,
    int M, int N, int K) {
    __shared__ bf16_t As[2][128 * 32];   // 2 x 8 KB
    __shared__ bf16_t Bs[2][64 * 32];    // 2 x 4 KB, N-major

    const int tid  = threadIdx.x;
    const int lane = tid & 31;
    const int wave = tid >> 5;        // 0..7
    const int wm   = wave >> 1;       // 0..3
    const int wn   = wave & 1;        // 0..1
    const int bm   = blockIdx.y * 128;
    const int bn   = blockIdx.x * 64;

    const unsigned as_base = (unsigned)(size_t)(void*)&As[0][0];
    const unsigned bs_base = (unsigned)(size_t)(void*)&Bs[0][0];

    v8f acc[2][2] = {};

    const int nk = K >> 5;
    gemm_stage(A, Wt, as_base, bs_base, bm, bn, K, 0, tid);

    for (int s = 0; s < nk; ++s) {
        const int buf = s & 1;
        const bool more = (s + 1 < nk);
        if (more) {
            gemm_stage(A, Wt, as_base + (unsigned)((buf ^ 1) * 8192),
                       bs_base + (unsigned)((buf ^ 1) * 4096),
                       bm, bn, K, (s + 1) * 32, tid);
            // drain only stage s (3 newer copies may stay in flight)
            asm volatile("s_wait_asynccnt 3" ::: "memory");
        } else {
            asm volatile("s_wait_asynccnt 0" ::: "memory");
        }
        __syncthreads();

        const bf16_t* as = &As[buf][0];
        const bf16_t* bs = &Bs[buf][0];
        v16bf afr[2], bfr[2];
        afr[0] = load_frag_a(as, 32, wm * 32, lane);
        afr[1] = load_frag_a(as, 32, wm * 32 + 16, lane);
        bfr[0] = load_frag_b_nmajor(bs, 32, wn * 32, lane);
        bfr[1] = load_frag_b_nmajor(bs, 32, wn * 32 + 16, lane);
#pragma unroll
        for (int i = 0; i < 2; ++i)
#pragma unroll
            for (int j = 0; j < 2; ++j)
                acc[i][j] = wmma_bf16(afr[i], bfr[j], acc[i][j]);
        __syncthreads();
    }

    // Writeout: D layout -> lane l, vgpr v: M = v + 8*(l>>4), N = l&15
    const int half = lane >> 4;
    const int nlo  = lane & 15;
#pragma unroll
    for (int i = 0; i < 2; ++i) {
#pragma unroll
        for (int j = 0; j < 2; ++j) {
            const int col = bn + wn * 32 + j * 16 + nlo;
            const float bv = bias ? bias[col] : 0.0f;
#pragma unroll
            for (int v = 0; v < 8; ++v) {
                const int row = bm + wm * 32 + i * 16 + v + half * 8;
                float val = acc[i][j][v] + bv;
                if (RELU) val = val > 0.0f ? val : 0.0f;
                if (OUT_BF16)
                    ((bf16_t*)C)[(size_t)row * N + col] = f2bf(val);
                else
                    ((float*)C)[(size_t)row * N + col] = val;
            }
        }
    }
}

// ---------------------------------------------------------------------------
// Fused sigmoid attention: one wave per (b, h, 16-query tile).
// ctx[b, q, h*64+d] = sum_k sigmoid(q.k/8 - log S) * V[k,d]   (bf16 out)
// Vt is depth-major: vt[(bh*64 + d)*S + s]
// ---------------------------------------------------------------------------
#define LDS_S 34   // 32 + pad

__global__ __launch_bounds__(32) void attn_kernel(
    const bf16_t* __restrict__ Q, const bf16_t* __restrict__ Km,
    const bf16_t* __restrict__ Vt, bf16_t* __restrict__ O, int S) {
    __shared__ bf16_t Sbf[16 * LDS_S];

    const int lane  = threadIdx.x;
    const int half  = lane >> 4;
    const int nlo   = lane & 15;
    const int bh    = blockIdx.y;
    const int b     = bh >> 3;
    const int h     = bh & 7;
    const int qbase = blockIdx.x * 16;
    const int DM    = 512;

    const size_t hoff = (size_t)(b * S) * DM + h * 64;
    const bf16_t* Qb  = Q + hoff;
    const bf16_t* Kb  = Km + hoff;
    const bf16_t* Vbt = Vt + (size_t)bh * 64 * S;

    const float SCALE = 0.125f;                 // 1/sqrt(64)
    const float SBIAS = -8.317766166719343f;    // -log(4096)

    // Q fragments: 16x64 -> two 16x32 A fragments (loaded once)
    v16bf qf[2];
#pragma unroll
    for (int kh = 0; kh < 2; ++kh) {
#pragma unroll
        for (int j = 0; j < 8; ++j) {
            const int dd = kh * 32 + akpat(j, half);
            const size_t rb = (size_t)(qbase + nlo) * DM;
            qf[kh][2 * j]     = Qb[rb + dd];
            qf[kh][2 * j + 1] = Qb[rb + dd + 1];
        }
    }

    v8f oacc[4] = {};

    for (int kbase = 0; kbase < S; kbase += 32) {
        // ---- scores: 16x32 block = two 16x16 WMMA tiles ----
#pragma unroll
        for (int nt = 0; nt < 2; ++nt) {
            v8f sc = {};
#pragma unroll
            for (int kh = 0; kh < 2; ++kh) {
                v16bf kf;  // B fragment of K^T (depth x keys)
                const size_t krow = (size_t)(kbase + nt * 16 + nlo) * DM;
#pragma unroll
                for (int j = 0; j < 8; ++j) {
                    const int dd = kh * 32 + bkpat(j, half);
                    kf[2 * j]     = Kb[krow + dd];
                    kf[2 * j + 1] = Kb[krow + dd + 1];
                }
                sc = wmma_bf16(qf[kh], kf, sc);
            }
            // sigmoid via v_exp + v_rcp; stash bf16 tile in LDS to re-swizzle
#pragma unroll
            for (int v = 0; v < 8; ++v) {
                const float xv = sc[v] * SCALE + SBIAS;
                const float p  = __builtin_amdgcn_rcpf(1.0f + __expf(-xv));
                const int mm = v + half * 8;      // query row 0..15
                const int nn = nt * 16 + nlo;     // key col 0..31
                Sbf[mm * LDS_S + nn] = f2bf(p);
            }
        }
        __syncthreads();

        // ---- out += S(16x32) @ V(32x64), V depth-major => contiguous ----
        const v16bf sfr = load_frag_a(Sbf, LDS_S, 0, lane);
#pragma unroll
        for (int dt = 0; dt < 4; ++dt) {
            const int dc = dt * 16 + nlo;
            const bf16_t* vrow = Vbt + (size_t)dc * S + kbase;
            v16bf vf;
#pragma unroll
            for (int j = 0; j < 8; ++j) {
                const int kk = bkpat(j, half);   // two contiguous runs of 8
                vf[2 * j]     = vrow[kk];
                vf[2 * j + 1] = vrow[kk + 1];
            }
            oacc[dt] = wmma_bf16(sfr, vf, oacc[dt]);
        }
        __syncthreads();
    }

    // write context in [B, S, 512] bf16 layout (heads concatenated)
#pragma unroll
    for (int dt = 0; dt < 4; ++dt) {
#pragma unroll
        for (int v = 0; v < 8; ++v) {
            const int row = qbase + v + half * 8;
            const int col = h * 64 + dt * 16 + nlo;
            O[((size_t)b * S + row) * DM + col] = f2bf(oacc[dt][v]);
        }
    }
}

// ---------------------------------------------------------------------------
// Fused residual + LayerNorm; optionally also emits a bf16 copy of the output
// ---------------------------------------------------------------------------
template <bool EMIT_BF16>
__global__ __launch_bounds__(256) void ln_kernel(
    const float* __restrict__ a, const float* __restrict__ r,
    const float* __restrict__ g, const float* __restrict__ be,
    float* __restrict__ out, bf16_t* __restrict__ outb) {
    __shared__ float ssum[8], ssq[8];
    __shared__ float s_mu, s_rstd;
    const int D = 512;
    const size_t row = blockIdx.x;
    const int t = threadIdx.x;

    const float v0 = a[row * D + t] + r[row * D + t];
    const float v1 = a[row * D + t + 256] + r[row * D + t + 256];
    float s = v0 + v1;
    float q = v0 * v0 + v1 * v1;
#pragma unroll
    for (int off = 16; off > 0; off >>= 1) {   // wave32 reduction
        s += __shfl_down(s, off, 32);
        q += __shfl_down(q, off, 32);
    }
    if ((t & 31) == 0) { ssum[t >> 5] = s; ssq[t >> 5] = q; }
    __syncthreads();
    if (t == 0) {
        float S = 0.0f, Q = 0.0f;
#pragma unroll
        for (int i = 0; i < 8; ++i) { S += ssum[i]; Q += ssq[i]; }
        const float mu = S * (1.0f / D);
        s_mu = mu;
        s_rstd = rsqrtf(Q * (1.0f / D) - mu * mu + 1e-5f);
    }
    __syncthreads();
    const float mu = s_mu, rs = s_rstd;
    const float o0 = (v0 - mu) * rs * g[t] + be[t];
    const float o1 = (v1 - mu) * rs * g[t + 256] + be[t + 256];
    out[row * D + t]       = o0;
    out[row * D + t + 256] = o1;
    if (EMIT_BF16) {
        outb[row * D + t]       = f2bf(o0);
        outb[row * D + t + 256] = f2bf(o1);
    }
}

// ---------------------------------------------------------------------------
// Launch
// ---------------------------------------------------------------------------
extern "C" void kernel_launch(void* const* d_in, const int* in_sizes, int n_in,
                              void* d_out, int out_size, void* d_ws, size_t ws_size,
                              hipStream_t stream) {
    const float* x       = (const float*)d_in[0];
    const float* Wq      = (const float*)d_in[1];
    const float* Wk      = (const float*)d_in[2];
    const float* Wv      = (const float*)d_in[3];
    const float* dense_w = (const float*)d_in[4];
    const float* dense_b = (const float*)d_in[5];
    const float* ff1_w   = (const float*)d_in[6];
    const float* ff1_b   = (const float*)d_in[7];
    const float* ff2_w   = (const float*)d_in[8];
    const float* ff2_b   = (const float*)d_in[9];
    const float* ln1_g   = (const float*)d_in[10];
    const float* ln1_b   = (const float*)d_in[11];
    const float* ln2_g   = (const float*)d_in[12];
    const float* ln2_b   = (const float*)d_in[13];
    float* out = (float*)d_out;

    const int ROWS = 8192;   // B*S
    const int D    = 512;
    const int F    = 2048;
    const int S    = 4096;

    // Workspace carve-up (~150 MB); weights stored transposed [N][K]
    char* p = (char*)d_ws;
    bf16_t* xb   = (bf16_t*)p; p += (size_t)ROWS * D * sizeof(bf16_t);
    bf16_t* wqt  = (bf16_t*)p; p += (size_t)D * D * sizeof(bf16_t);
    bf16_t* wkt  = (bf16_t*)p; p += (size_t)D * D * sizeof(bf16_t);
    bf16_t* wvt  = (bf16_t*)p; p += (size_t)D * D * sizeof(bf16_t);
    bf16_t* dwt  = (bf16_t*)p; p += (size_t)D * D * sizeof(bf16_t);
    bf16_t* f1wt = (bf16_t*)p; p += (size_t)D * F * sizeof(bf16_t);
    bf16_t* f2wt = (bf16_t*)p; p += (size_t)F * D * sizeof(bf16_t);
    bf16_t* qb   = (bf16_t*)p; p += (size_t)ROWS * D * sizeof(bf16_t);
    bf16_t* kb   = (bf16_t*)p; p += (size_t)ROWS * D * sizeof(bf16_t);
    bf16_t* vb   = (bf16_t*)p; p += (size_t)ROWS * D * sizeof(bf16_t);
    bf16_t* vt   = (bf16_t*)p; p += (size_t)ROWS * D * sizeof(bf16_t);
    bf16_t* ctxb = (bf16_t*)p; p += (size_t)ROWS * D * sizeof(bf16_t);
    bf16_t* x1b  = (bf16_t*)p; p += (size_t)ROWS * D * sizeof(bf16_t);
    bf16_t* h1   = (bf16_t*)p; p += (size_t)ROWS * F * sizeof(bf16_t);
    float*  prj  = (float*)p;  p += (size_t)ROWS * D * sizeof(float);
    float*  x1   = (float*)p;  p += (size_t)ROWS * D * sizeof(float);
    float*  f2   = (float*)p;  p += (size_t)ROWS * D * sizeof(float);

    const dim3 blk(256);
    const dim3 gD(D / 64, ROWS / 128);   // N=512 GEMMs
    const dim3 gF(F / 64, ROWS / 128);   // N=2048 GEMM

    // One-time conversions (weights: transpose to [N][K])
    cvt_kernel<<<dim3(ROWS * D / 1024), blk, 0, stream>>>(x, xb, ROWS * D);
    cvt_t_kernel<<<dim3(D / 64, D / 64), blk, 0, stream>>>(Wq, wqt, D, D);
    cvt_t_kernel<<<dim3(D / 64, D / 64), blk, 0, stream>>>(Wk, wkt, D, D);
    cvt_t_kernel<<<dim3(D / 64, D / 64), blk, 0, stream>>>(Wv, wvt, D, D);
    cvt_t_kernel<<<dim3(D / 64, D / 64), blk, 0, stream>>>(dense_w, dwt, D, D);
    cvt_t_kernel<<<dim3(D / 64, F / 64), blk, 0, stream>>>(ff1_w, f1wt, D, F);
    cvt_t_kernel<<<dim3(F / 64, D / 64), blk, 0, stream>>>(ff2_w, f2wt, F, D);

    // Q/K/V projections (bf16 out)
    gemm_kernel<true, false><<<gD, blk, 0, stream>>>(xb, wqt, nullptr, qb, ROWS, D, D);
    gemm_kernel<true, false><<<gD, blk, 0, stream>>>(xb, wkt, nullptr, kb, ROWS, D, D);
    gemm_kernel<true, false><<<gD, blk, 0, stream>>>(xb, wvt, nullptr, vb, ROWS, D, D);

    // V -> depth-major layout for contiguous B-fragments in attention
    vtrans_kernel<<<dim3(S / 64, 16), blk, 0, stream>>>(vb, vt, S);

    // Fused sigmoid attention (bf16 context out)
    attn_kernel<<<dim3(S / 16, 16), dim3(32), 0, stream>>>(qb, kb, vt, ctxb, S);

    // Dense projection + LN1 (LN1 also emits bf16 copy for FF1)
    gemm_kernel<false, false><<<gD, blk, 0, stream>>>(ctxb, dwt, dense_b, prj, ROWS, D, D);
    ln_kernel<true><<<dim3(ROWS), blk, 0, stream>>>(x, prj, ln1_g, ln1_b, x1, x1b);

    // FFN + LN2
    gemm_kernel<true, true><<<gF, blk, 0, stream>>>(x1b, f1wt, ff1_b, h1, ROWS, F, D);
    gemm_kernel<false, false><<<gD, blk, 0, stream>>>(h1, f2wt, ff2_b, f2, ROWS, D, F);
    ln_kernel<false><<<dim3(ROWS), blk, 0, stream>>>(x1, f2, ln2_g, ln2_b, out, nullptr);
}
